// FusedMoE_55482387530504
// MI455X (gfx1250) — matrix operand order repaired and checked by reading, not compiled, not run on hardware
//
#include <hip/hip_runtime.h>
#include <hip/hip_bf16.h>
#include <cstdint>
#include <cstddef>

// ---------------------------------------------------------------------------
// Fused MoE (top-2 of 8 experts), T=8192 H=2048 E=8 I=5632, f32 in/out.
// Compute-bound (AI ~950 FLOP/B) -> bf16 WMMA (v_wmma_f32_16x16x32_bf16),
// f32 accumulation. Weights stream f32 from HBM/L2, converted to bf16
// in-register during LDS staging (avoids +1.6GB of convert-to-scratch traffic;
// per-expert w13 = 92MB fits the 192MB L2 for tile re-reads).
// Pipeline: router -> segment setup -> fill -> GEMM1(+silu fuse) -> GEMM2(+atomic scatter)
// Both GEMMs: double-buffered LDS, ONE s_barrier per K-step.
// GEMM2 A-tile (already bf16) staged by the Tensor Data Mover
// (tensor_load_to_lds, 6-arg toolchain form + s_wait_tensorcnt), with TDM pad
// config matching the 80B bank-spread LDS row stride.
// Workspace requirement: ~197 MB (dominated by bf16 h-scratch CAP x I).
// ---------------------------------------------------------------------------

constexpr int T_ = 8192;
constexpr int H_ = 2048;
constexpr int E_ = 8;
constexpr int I_ = 5632;

constexpr int BM  = 128;
constexpr int BN  = 128;
constexpr int BK  = 32;
constexpr int LDA = BK + 8;            // LDS row stride (bf16 elems) = 80B, bank-spread
constexpr int CAP = 17408;             // 16384 entries + 8*128 worst-case pad
constexpr int MAXTILES = CAP / BM;     // 136 static row tiles

typedef __attribute__((ext_vector_type(16))) __bf16 v16bf;
typedef __attribute__((ext_vector_type(8)))  float  v8f;
typedef __attribute__((ext_vector_type(8)))  __bf16 bf16x8;
typedef __attribute__((ext_vector_type(4)))  float  f32x4;
typedef __attribute__((ext_vector_type(4)))  unsigned int u32x4;
typedef __attribute__((ext_vector_type(8)))  int    i32x8;
typedef __attribute__((ext_vector_type(4)))  int    i32x4;

#if defined(__has_builtin)
#  if __has_builtin(__builtin_amdgcn_tensor_load_to_lds) && \
      __has_builtin(__builtin_amdgcn_s_wait_tensorcnt)
#    define USE_TDM 1
#  endif
#endif
#ifndef USE_TDM
#  define USE_TDM 0
#endif

union FragAB { v16bf v; bf16x8 h[2]; };

__device__ __forceinline__ bf16x8 cvt8(f32x4 a, f32x4 b) {
  bf16x8 r;
  r[0] = (__bf16)a[0]; r[1] = (__bf16)a[1]; r[2] = (__bf16)a[2]; r[3] = (__bf16)a[3];
  r[4] = (__bf16)b[0]; r[5] = (__bf16)b[1]; r[6] = (__bf16)b[2]; r[7] = (__bf16)b[3];
  return r;
}

__device__ __forceinline__ v8f wmma_bf16(v16bf a, v16bf b, v8f c) {
  return __builtin_amdgcn_wmma_f32_16x16x32_bf16(false, a, false, b, (short)0, c,
                                                 false, false);
}

// A-matrix 16x32 bf16 fragment (ISA 7.12.2): lane = {m = L%16, khalf = L/16},
// VGPR0-3 hold K = khalf*8 + 0..7, VGPR4-7 hold K = 16 + khalf*8 + 0..7.
__device__ __forceinline__ v16bf frag_a(const __bf16* tile, int m, int khalf) {
  FragAB f;
  f.h[0] = *(const bf16x8*)(tile + m * LDA + khalf * 8);
  f.h[1] = *(const bf16x8*)(tile + m * LDA + 16 + khalf * 8);
  return f.v;
}

// B-matrix 32x16 bf16 fragment: lane = {n = L%16, khalf = L/16},
// VGPR0-3 hold K = khalf*8 + 0..7, VGPR4-7 hold K = 16 + khalf*8 + 0..7.
__device__ __forceinline__ v16bf frag_b(const __bf16* tile, int n, int khalf) {
  FragAB f;
  f.h[0] = *(const bf16x8*)(tile + n * LDA + khalf * 8);
  f.h[1] = *(const bf16x8*)(tile + n * LDA + 16 + khalf * 8);
  return f.v;
}

__device__ __forceinline__ void atomic_add_f32(float* p, float v) {
  unsafeAtomicAdd(p, v);   // -> global_atomic_add_f32
}

#if USE_TDM
// TDM descriptor for a 2D bf16 tile: 128 rows x 32 cols, row stride I_ elems
// in global, landed in LDS with 16B padding after every 64B so rows sit at
// the 80B (LDA) stride the fragment loaders expect. 2D tile -> descriptor
// groups 2/3 (and the trailing i32x8 group of the 6-arg builtin) are unused.
__device__ __forceinline__ void tdm_load_tile(const __bf16* gsrc, __bf16* ldst,
                                              int rows_avail) {
  const unsigned lds_addr = (unsigned)(uintptr_t)ldst;     // low 32b == LDS offset
  const unsigned long long ga = (unsigned long long)(uintptr_t)gsrc;
  u32x4 g0;
  g0[0] = 1u;                                             // count=1, user D#, no gather
  g0[1] = lds_addr;                                       // lds_addr [63:32]
  g0[2] = (unsigned)ga;                                   // global_addr lo
  g0[3] = ((unsigned)(ga >> 32) & 0x01FFFFFFu) | (2u << 30);  // addr hi | type=2
  i32x8 g1;
  g1[0] = (1 << 16)      // data_size = 2 bytes
        | (1 << 20)      // pad_enable
        | (3 << 22)      // pad_interval: 16 DWORDs (64B) between pads
        | (3 << 25);     // pad_amount:   4 DWORDs (16B)
  g1[1] = (int)(((unsigned)I_ & 0xFFFFu) << 16);          // tensor_dim0 lo16 @ [63:48]
  g1[2] = (int)((((unsigned)I_ >> 16) & 0xFFFFu)          // tensor_dim0 hi16
        | (((unsigned)rows_avail & 0xFFFFu) << 16));      // tensor_dim1 lo16
  g1[3] = (int)((((unsigned)rows_avail >> 16) & 0xFFFFu)  // tensor_dim1 hi16
        | (32u << 16));                                   // tile_dim0 = 32 elems
  g1[4] = 128;                                            // tile_dim1 = 128 rows
  g1[5] = I_;                                             // tensor_dim0_stride lo
  g1[6] = 0;
  g1[7] = 0;
  const i32x4 z4 = {0, 0, 0, 0};
  const i32x8 z8 = {0, 0, 0, 0, 0, 0, 0, 0};
  __builtin_amdgcn_tensor_load_to_lds(g0, g1, z4, z4, z8, 0);
}
#endif

// ---------------------------------------------------------------------------
// Kernel 0: zero the routing counters.
// ---------------------------------------------------------------------------
__global__ void moe_zero_counters(int* counts, int* count2) {
  int i = threadIdx.x;
  if (i < E_) { counts[i] = 0; count2[i] = 0; }
}

// ---------------------------------------------------------------------------
// Kernel 1: router. Top-2 of 8 logits; renormalized softmax weights reduce to
// wa = 1/(1+exp(lb-la)) since the softmax denominator cancels.
// ---------------------------------------------------------------------------
__global__ void moe_router(const float* __restrict__ logits,
                           int* __restrict__ topk_idx, float* __restrict__ topk_w,
                           int* __restrict__ counts) {
  const int t = blockIdx.x * blockDim.x + threadIdx.x;
  if (t >= T_) return;
  float l[E_];
#pragma unroll
  for (int e = 0; e < E_; ++e) l[e] = logits[t * E_ + e];
  int i0 = 0; float v0 = l[0];
#pragma unroll
  for (int e = 1; e < E_; ++e) if (l[e] > v0) { v0 = l[e]; i0 = e; }
  int i1 = -1; float v1 = -3.4e38f;
#pragma unroll
  for (int e = 0; e < E_; ++e) if (e != i0 && l[e] > v1) { v1 = l[e]; i1 = e; }
  const float w0 = 1.f / (1.f + __expf(v1 - v0));
  const float w1 = 1.f - w0;
  topk_idx[2 * t + 0] = i0;  topk_w[2 * t + 0] = w0;
  topk_idx[2 * t + 1] = i1;  topk_w[2 * t + 1] = w1;
  atomicAdd(&counts[i0], 1);
  atomicAdd(&counts[i1], 1);
}

// ---------------------------------------------------------------------------
// Kernel 2: segment setup (single thread; E=8, 136 tiles -> trivial).
// ---------------------------------------------------------------------------
__global__ void moe_setup(const int* __restrict__ counts, int* __restrict__ count2,
                          int* __restrict__ seg_off, int* __restrict__ row_expert) {
  if (threadIdx.x != 0 || blockIdx.x != 0) return;
  int off = 0;
  for (int e = 0; e < E_; ++e) {
    seg_off[e] = off;
    off += (counts[e] + BM - 1) & ~(BM - 1);
    count2[e] = 0;
  }
  seg_off[E_] = off;
  for (int tl = 0; tl < MAXTILES; ++tl) row_expert[tl] = -1;
  for (int e = 0; e < E_; ++e)
    for (int tl = seg_off[e] / BM; tl < seg_off[e + 1] / BM; ++tl) row_expert[tl] = e;
}

// ---------------------------------------------------------------------------
// Kernel 3: init entry rows (padding rows stay token = -1 => masked).
// ---------------------------------------------------------------------------
__global__ void moe_init_entries(int* __restrict__ etok, float* __restrict__ ew) {
  const int i = blockIdx.x * blockDim.x + threadIdx.x;
  if (i < CAP) { etok[i] = -1; ew[i] = 0.f; }
}

// ---------------------------------------------------------------------------
// Kernel 4: zero the f32 output (GEMM2 accumulates atomically).
// ---------------------------------------------------------------------------
__global__ void moe_zero_out(float* __restrict__ out) {
  const size_t i = (size_t)blockIdx.x * blockDim.x + threadIdx.x;
  if (i < (size_t)T_ * H_ / 4) ((f32x4*)out)[i] = (f32x4){0.f, 0.f, 0.f, 0.f};
}

// ---------------------------------------------------------------------------
// Kernel 5: scatter (token,k) assignments into per-expert row segments.
// ---------------------------------------------------------------------------
__global__ void moe_fill(const int* __restrict__ topk_idx, const float* __restrict__ topk_w,
                         const int* __restrict__ seg_off, int* __restrict__ count2,
                         int* __restrict__ etok, float* __restrict__ ew) {
  const int i = blockIdx.x * blockDim.x + threadIdx.x;
  if (i >= T_ * 2) return;
  const int e = topk_idx[i];
  const int pos = atomicAdd(&count2[e], 1);
  const int row = seg_off[e] + pos;
  etok[row] = i >> 1;
  ew[row]   = topk_w[i];
}

// ---------------------------------------------------------------------------
// Kernel 6: GEMM1. h[row, 0:I] = silu(Xg @ W13g^T) * (Xg @ W13u^T), bf16 out.
// 128x128 tile, BK=32, 8 waves (2x4), gate+up accumulated together.
// Double-buffered LDS, one barrier per K-step; f32->bf16 convert in-register.
// ---------------------------------------------------------------------------
__global__ __launch_bounds__(256) void moe_gemm1(
    const float* __restrict__ x, const float* __restrict__ w13,
    const int* __restrict__ entry_token, const int* __restrict__ row_expert,
    __bf16* __restrict__ hbuf) {
  __shared__ __bf16 As[2][BM * LDA];
  __shared__ __bf16 Bgs[2][BN * LDA];
  __shared__ __bf16 Bus[2][BN * LDA];
  __shared__ int    toks[BM];

  const int e = row_expert[blockIdx.y];
  if (e < 0) return;
  const int row0 = blockIdx.y * BM;
  const int nb0  = blockIdx.x * BN;
  const int tid  = threadIdx.x;

  if (tid < BM) toks[tid] = entry_token[row0 + tid];
  __syncthreads();

  // Loader role: thread -> (row, 16-float half of the 32-wide K slab).
  const int lr = tid >> 1;
  const int hf = tid & 1;
  const int tok = toks[lr];
  const float* xa = (tok >= 0) ? (x + (size_t)tok * H_ + hf * 16) : nullptr;
  const float* wg = w13 + ((size_t)e * 2 * I_ + nb0 + lr) * (size_t)H_ + hf * 16;
  const float* wu = wg + (size_t)I_ * H_;

  // Compute role: 2x4 wave grid, each wave owns 64x32 of the 128x128 tile.
  const int lane = tid & 31;
  const int wv   = tid >> 5;
  const int mw = wv >> 2, nw = wv & 3;
  const int lhi = lane >> 4, llo = lane & 15;

  f32x4 ax0, ax1, ax2, ax3, bg0, bg1, bg2, bg3, bu0, bu1, bu2, bu3;

  auto load_tiles = [&](int kb) {
    ax0 = (f32x4){0.f, 0.f, 0.f, 0.f}; ax1 = ax0; ax2 = ax0; ax3 = ax0;
    if (xa) {
      const f32x4* s = (const f32x4*)(xa + kb);
      ax0 = s[0]; ax1 = s[1]; ax2 = s[2]; ax3 = s[3];
    }
    const f32x4* sg = (const f32x4*)(wg + kb);
    bg0 = sg[0]; bg1 = sg[1]; bg2 = sg[2]; bg3 = sg[3];
    const f32x4* su = (const f32x4*)(wu + kb);
    bu0 = su[0]; bu1 = su[1]; bu2 = su[2]; bu3 = su[3];
  };
  auto store_tiles = [&](int b) {
    bf16x8* d = (bf16x8*)(As[b] + lr * LDA + hf * 16);
    d[0] = cvt8(ax0, ax1); d[1] = cvt8(ax2, ax3);
    d = (bf16x8*)(Bgs[b] + lr * LDA + hf * 16);
    d[0] = cvt8(bg0, bg1); d[1] = cvt8(bg2, bg3);
    d = (bf16x8*)(Bus[b] + lr * LDA + hf * 16);
    d[0] = cvt8(bu0, bu1); d[1] = cvt8(bu2, bu3);
  };

  v8f accG[4][2] = {};
  v8f accU[4][2] = {};

  load_tiles(0);
  store_tiles(0);
  __syncthreads();

  int cur = 0;
  for (int kb = 0; kb < H_; kb += BK) {
    const bool nxt = (kb + BK) < H_;
    if (nxt) {
      if (kb + 4 * BK < H_) {               // never cross the row end
        __builtin_prefetch(wg + kb + 4 * BK, 0, 0);
        __builtin_prefetch(wu + kb + 4 * BK, 0, 0);
      }
      load_tiles(kb + BK);
    }

    v16bf af[4];
#pragma unroll
    for (int mi = 0; mi < 4; ++mi)
      af[mi] = frag_a(As[cur], mw * 64 + mi * 16 + llo, lhi);
#pragma unroll
    for (int ni = 0; ni < 2; ++ni) {
      const v16bf bgf = frag_b(Bgs[cur], nw * 32 + ni * 16 + llo, lhi);
      const v16bf buf = frag_b(Bus[cur], nw * 32 + ni * 16 + llo, lhi);
#pragma unroll
      for (int mi = 0; mi < 4; ++mi) {
        accG[mi][ni] = wmma_bf16(af[mi], bgf, accG[mi][ni]);
        accU[mi][ni] = wmma_bf16(af[mi], buf, accU[mi][ni]);
      }
    }

    if (nxt) store_tiles(cur ^ 1);
    __syncthreads();
    cur ^= 1;
  }

  // Fused epilogue: h = silu(gate) * up, bf16 scratch.
  // C layout: lane -> n = L%16; vgpr r -> m = r + 8*(L/16).
#pragma unroll
  for (int mi = 0; mi < 4; ++mi)
#pragma unroll
    for (int ni = 0; ni < 2; ++ni)
#pragma unroll
      for (int r = 0; r < 8; ++r) {
        const int m   = mw * 64 + mi * 16 + r + 8 * lhi;
        const int col = nb0 + nw * 32 + ni * 16 + llo;
        const float g = accG[mi][ni][r];
        const float u = accU[mi][ni][r];
        const float hv = (g / (1.f + __expf(-g))) * u;
        hbuf[(size_t)(row0 + m) * I_ + col] = (__bf16)hv;
      }
}

// ---------------------------------------------------------------------------
// Kernel 7: GEMM2. out[token] += w_route * (h_row @ W2^T), atomic f32 scatter.
// A-tile (bf16 h scratch) staged by TDM when available; B converted f32->bf16.
// ---------------------------------------------------------------------------
__global__ __launch_bounds__(256) void moe_gemm2(
    const __bf16* __restrict__ hbuf, const float* __restrict__ w2,
    const int* __restrict__ entry_token, const float* __restrict__ entry_w,
    const int* __restrict__ row_expert, float* __restrict__ out) {
  __shared__ __bf16 As[2][BM * LDA];
  __shared__ __bf16 Bs[2][BN * LDA];
  __shared__ int    toks[BM];
  __shared__ float  wrow[BM];

  const int e = row_expert[blockIdx.y];
  if (e < 0) return;
  const int row0 = blockIdx.y * BM;
  const int hb0  = blockIdx.x * BN;
  const int tid  = threadIdx.x;

  if (tid < BM) {
    toks[tid] = entry_token[row0 + tid];
    wrow[tid] = entry_w[row0 + tid];
  }
  __syncthreads();

  const int lr = tid >> 1;
  const int hf = tid & 1;
  const __bf16* ha = hbuf + (size_t)(row0 + lr) * I_ + hf * 16;
  const float*  wb = w2 + ((size_t)e * H_ + hb0 + lr) * (size_t)I_ + hf * 16;

  const int lane = tid & 31;
  const int wv   = tid >> 5;
  const int mw = wv >> 2, nw = wv & 3;
  const int lhi = lane >> 4, llo = lane & 15;

  bf16x8 a0, a1;
  f32x4 b0, b1, b2, b3;

  auto loadB = [&](int kb) {
    const f32x4* s = (const f32x4*)(wb + kb);
    b0 = s[0]; b1 = s[1]; b2 = s[2]; b3 = s[3];
  };
  auto storeB = [&](int b) {
    bf16x8* d = (bf16x8*)(Bs[b] + lr * LDA + hf * 16);
    d[0] = cvt8(b0, b1); d[1] = cvt8(b2, b3);
  };
#if !USE_TDM
  auto loadA = [&](int kb) {
    const bf16x8* s = (const bf16x8*)(ha + kb);
    a0 = s[0]; a1 = s[1];
  };
  auto storeA = [&](int b) {
    bf16x8* d = (bf16x8*)(As[b] + lr * LDA + hf * 16);
    d[0] = a0; d[1] = a1;
  };
#endif

  v8f acc[4][2] = {};

  // ---- prologue: stage K-slab 0 into buffer 0
#if USE_TDM
  if (wv == 0)
    tdm_load_tile(hbuf + (size_t)row0 * I_, As[0], CAP - row0);
#else
  loadA(0); storeA(0);
#endif
  loadB(0); storeB(0);
#if USE_TDM
  if (wv == 0) __builtin_amdgcn_s_wait_tensorcnt(0);
#endif
  __syncthreads();

  int cur = 0;
  for (int kb = 0; kb < I_; kb += BK) {
    const bool nxt = (kb + BK) < I_;
    if (nxt) {
      if (kb + 4 * BK < I_) __builtin_prefetch(wb + kb + 4 * BK, 0, 0);
#if USE_TDM
      if (wv == 0)
        tdm_load_tile(hbuf + (size_t)row0 * I_ + kb + BK, As[cur ^ 1], CAP - row0);
#else
      loadA(kb + BK);
#endif
      loadB(kb + BK);
    }

    v16bf af[4];
#pragma unroll
    for (int mi = 0; mi < 4; ++mi)
      af[mi] = frag_a(As[cur], mw * 64 + mi * 16 + llo, lhi);
#pragma unroll
    for (int ni = 0; ni < 2; ++ni) {
      const v16bf b = frag_b(Bs[cur], nw * 32 + ni * 16 + llo, lhi);
#pragma unroll
      for (int mi = 0; mi < 4; ++mi)
        acc[mi][ni] = wmma_bf16(af[mi], b, acc[mi][ni]);
    }

    if (nxt) {
      storeB(cur ^ 1);
#if !USE_TDM
      storeA(cur ^ 1);
#endif
#if USE_TDM
      if (wv == 0) __builtin_amdgcn_s_wait_tensorcnt(0);
#endif
    }
    __syncthreads();
    cur ^= 1;
  }

#pragma unroll
  for (int mi = 0; mi < 4; ++mi)
#pragma unroll
    for (int ni = 0; ni < 2; ++ni)
#pragma unroll
      for (int r = 0; r < 8; ++r) {
        const int m = mw * 64 + mi * 16 + r + 8 * lhi;
        const int t = toks[m];
        if (t >= 0) {
          const int col = hb0 + nw * 32 + ni * 16 + llo;
          atomic_add_f32(out + (size_t)t * H_ + col, acc[mi][ni][r] * wrow[m]);
        }
      }
}

// ---------------------------------------------------------------------------
// Launch
// ---------------------------------------------------------------------------
extern "C" void kernel_launch(void* const* d_in, const int* in_sizes, int n_in,
                              void* d_out, int out_size, void* d_ws, size_t ws_size,
                              hipStream_t stream) {
  (void)in_sizes; (void)n_in; (void)out_size; (void)ws_size;
  const float* x      = (const float*)d_in[0];
  const float* logits = (const float*)d_in[1];
  const float* w13    = (const float*)d_in[2];
  const float* w2     = (const float*)d_in[3];
  float* out = (float*)d_out;

  char* ws = (char*)d_ws;
  size_t off = 0;
  auto alloc = [&](size_t bytes) -> void* {
    void* p = (void*)(ws + off);
    off += (bytes + 255) & ~(size_t)255;
    return p;
  };
  int*    topk_idx = (int*)   alloc((size_t)T_ * 2 * sizeof(int));
  float*  topk_w   = (float*) alloc((size_t)T_ * 2 * sizeof(float));
  int*    counts   = (int*)   alloc(E_ * sizeof(int));
  int*    count2   = (int*)   alloc(E_ * sizeof(int));
  int*    seg_off  = (int*)   alloc((E_ + 1) * sizeof(int));
  int*    row_exp  = (int*)   alloc(MAXTILES * sizeof(int));
  int*    etok     = (int*)   alloc(CAP * sizeof(int));
  float*  ew       = (float*) alloc(CAP * sizeof(float));
  __bf16* hbuf     = (__bf16*)alloc((size_t)CAP * I_ * sizeof(__bf16));  // ~196 MB

  moe_zero_counters<<<1, 64, 0, stream>>>(counts, count2);
  moe_router<<<(T_ + 255) / 256, 256, 0, stream>>>(logits, topk_idx, topk_w, counts);
  moe_setup<<<1, 1, 0, stream>>>(counts, count2, seg_off, row_exp);
  moe_init_entries<<<(CAP + 255) / 256, 256, 0, stream>>>(etok, ew);
  moe_zero_out<<<(int)(((size_t)T_ * H_ / 4 + 255) / 256), 256, 0, stream>>>(out);
  moe_fill<<<(T_ * 2 + 255) / 256, 256, 0, stream>>>(topk_idx, topk_w, seg_off,
                                                     count2, etok, ew);
  dim3 g1(I_ / BN, MAXTILES);
  moe_gemm1<<<g1, 256, 0, stream>>>(x, w13, etok, row_exp, hbuf);
  dim3 g2(H_ / BN, MAXTILES);
  moe_gemm2<<<g2, 256, 0, stream>>>(hbuf, w2, etok, ew, row_exp, out);
}
